// MultiHeadAttention_33595234189826
// MI455X (gfx1250) — matrix operand order repaired and checked by reading, not compiled, not run on hardware
//
#include <hip/hip_runtime.h>
#include <hip/hip_bf16.h>

typedef __attribute__((ext_vector_type(16))) _Float16 v16h;
typedef __attribute__((ext_vector_type(8)))  float    v8f;
typedef __attribute__((ext_vector_type(4)))  unsigned int v4u;
typedef __attribute__((ext_vector_type(8)))  int      v8i_;
typedef __attribute__((ext_vector_type(4)))  int      v4i;

#define B_ 2
#define S_ 2048
#define E_ 1024
#define H_ 16
#define D_ 64

#if defined(__HIP_DEVICE_COMPILE__) && \
    __has_builtin(__builtin_amdgcn_tensor_load_to_lds) && \
    __has_builtin(__builtin_amdgcn_s_wait_tensorcnt)
#define TDM_OK 1
#else
#define TDM_OK 0
#endif

// A-fragment K index for V_WMMA_F32_16X16X32_F16 (16x32 f16 A):
// lanes 0-15: e0..7 -> K0..7,  e8..15 -> K16..23
// lanes16-31: e0..7 -> K8..15, e8..15 -> K24..31
__device__ __forceinline__ int a_kmap(int lane, int e) {
    int khalf = (lane >> 4) * 8;
    return (e < 8) ? (khalf + e) : (16 + khalf + (e - 8));
}
// B-fragment (32x16 f16 B), column (lane&15), same K order as A:
// lanes 0-15: e0..7 -> K0..7, e8..15 -> K16..23; lanes 16-31: K8..15, K24..31.
__device__ __forceinline__ int b_kmap(int lane, int e) {
    int khalf = (lane >> 4) * 8;
    return (e < 8) ? (khalf + e) : (16 + khalf + (e - 8));
}

__device__ __forceinline__ v8f wmma_f16(v16h a, v16h b, v8f c) {
    return __builtin_amdgcn_wmma_f32_16x16x32_f16(false, a, false, b,
                                                  (short)0, c, false, false);
}

#if TDM_OK
// LDS byte offset of a __shared__ object (flat LDS aperture keeps the
// offset in the low 32 bits; upper bits are the aperture base).
__device__ __forceinline__ unsigned lds_off_of(const void* p) {
    return (unsigned)(unsigned long long)p;
}

// TDM: load a 2-D tile (tile1 rows of tile0 elements, row stride stride0
// elements) from global into LDS. Descriptor per CDNA5 ISA D# spec:
//   group0: count=1 | lds_addr | global_addr[56:0] | type=2
//   group1: data_size, tensor dims (generous, no OOB clamp), tile dims,
//           tensor_dim0_stride
//   groups 2/3: zero (<= 2-D tensor)
__device__ __forceinline__ void tdm_load_tile(unsigned lds_off, const void* gptr,
                                              unsigned ds_log,   // 0:1B 1:2B 2:4B 3:8B
                                              unsigned tile0, unsigned tile1,
                                              unsigned stride0) {
    unsigned long long ga = (unsigned long long)gptr;
    v4u g0;
    g0.x = 1u;                                                  // count=1 (user D#)
    g0.y = lds_off;                                             // lds_addr
    g0.z = (unsigned)ga;                                        // global_addr[31:0]
    g0.w = (unsigned)((ga >> 32) & 0x01FFFFFFu) | (2u << 30);   // [56:32] | type=2
    const unsigned td0 = (stride0 > tile0) ? stride0 : tile0;   // tensor_dim0
    const unsigned td1 = 0x40000000u;                           // huge tensor_dim1
    v8i_ g1;
    g1[0] = (int)(ds_log << 16);                          // wg_mask=0 | data_size
    g1[1] = (int)((td0 & 0xFFFFu) << 16);                 // bar_addr=0 | dim0 lo16
    g1[2] = (int)((td0 >> 16) | ((td1 & 0xFFFFu) << 16)); // dim0 hi16 | dim1 lo16
    g1[3] = (int)((td1 >> 16) | (tile0 << 16));           // dim1 hi16 | tile_dim0
    g1[4] = (int)(tile1 & 0xFFFFu);                       // tile_dim1 | tile_dim2=0
    g1[5] = (int)stride0;                                 // tensor_dim0_stride lo32
    g1[6] = 0;                                            // stride0 hi | dim1_stride
    g1[7] = 0;
    v4i z4 = {};
#if defined(__clang_major__) && __clang_major__ >= 23
    v8i_ z8 = {};
    __builtin_amdgcn_tensor_load_to_lds(g0, g1, z4, z4, z8, 0);
#else
    __builtin_amdgcn_tensor_load_to_lds(g0, g1, z4, z4, 0);
#endif
}
#endif  // TDM_OK

// ---------------------------------------------------------------------------
// Kernel 1: per-head projection  out[bh,s,d] = x[b,s,:] @ W[h,:,d]  (f32->f16)
// grid (S/64, B*H), block 128 (4 waves). K-step tiles staged in LDS via TDM:
//   X tile: 64 rows x 32 f32 (row stride E), W tile: contiguous 8 KB.
// ---------------------------------------------------------------------------
__global__ void mha_proj_kernel(const float* __restrict__ x,
                                const float* __restrict__ W,
                                _Float16* __restrict__ out) {
    __shared__ float Xlds[64 * 32];
    __shared__ float Wlds[32 * 64];
    const int tid   = threadIdx.x;
    const int lane  = tid & 31;
    const int wave  = tid >> 5;
    const int mrow0 = blockIdx.x * 64;
    const int mbase = mrow0 + wave * 16;
    const int bh    = blockIdx.y;
    const int b = bh / H_, h = bh % H_;
    const int ncol = lane & 15;
    const int arow = lane & 15;
    const float* Wh = W + (size_t)h * E_ * D_;

    v8f zero = {};
    v8f acc[4];
    #pragma unroll
    for (int nt = 0; nt < 4; ++nt) acc[nt] = zero;

    for (int k0 = 0; k0 < E_; k0 += 32) {
        __syncthreads();
#if TDM_OK
        if (wave == 0) {
            tdm_load_tile(lds_off_of(Xlds), &x[(size_t)(b * S_ + mrow0) * E_ + k0],
                          2u, 32u, 64u, (unsigned)E_);          // 64 rows x 32 f32
            tdm_load_tile(lds_off_of(Wlds), &Wh[(size_t)k0 * D_],
                          3u, 1024u, 1u, 1024u);                // contiguous 8 KB
            __builtin_amdgcn_s_wait_tensorcnt(0);
        }
#else
        for (int i = tid; i < 64 * 32; i += 128)
            Xlds[i] = x[(size_t)(b * S_ + mrow0 + (i >> 5)) * E_ + k0 + (i & 31)];
        for (int i = tid; i < 32 * 64; i += 128)
            Wlds[i] = Wh[(size_t)(k0 + (i >> 6)) * D_ + (i & 63)];
#endif
        __syncthreads();
        v16h a;
        #pragma unroll
        for (int e = 0; e < 16; ++e)
            a[e] = (_Float16)Xlds[(wave * 16 + arow) * 32 + a_kmap(lane, e)];
        #pragma unroll
        for (int nt = 0; nt < 4; ++nt) {
            v16h bf;
            #pragma unroll
            for (int e = 0; e < 16; ++e)
                bf[e] = (_Float16)Wlds[b_kmap(lane, e) * 64 + nt * 16 + ncol];
            acc[nt] = wmma_f16(a, bf, acc[nt]);
        }
    }
    const int mh = (lane >> 4) * 8;          // C layout: VGPR r -> M = mh + r
    #pragma unroll
    for (int nt = 0; nt < 4; ++nt)
        #pragma unroll
        for (int r = 0; r < 8; ++r)
            out[((size_t)(bh * S_ + mbase + mh + r)) * D_ + nt * 16 + ncol] =
                (_Float16)acc[nt][r];
}

// ---------------------------------------------------------------------------
// Kernel 2: causal flash attention, one wave per 16-query tile per (b,h).
// K/V 32-key tiles (contiguous 4 KB each) staged in LDS via TDM; online
// softmax; exp(P) bounced through LDS (C-layout -> A-layout) for P @ V.
// ---------------------------------------------------------------------------
__global__ void mha_attn_kernel(const _Float16* __restrict__ Q,
                                const _Float16* __restrict__ K,
                                const _Float16* __restrict__ V,
                                _Float16* __restrict__ attn) {
    __shared__ _Float16 Klds[32 * D_];
    __shared__ _Float16 Vlds[32 * D_];
    __shared__ _Float16 Plds[16 * 32];
    const int lane  = threadIdx.x & 31;
    const int qbase = blockIdx.x * 16;
    const int bh    = blockIdx.y;
    const int b = bh / H_, h = bh % H_;
    const _Float16* Qbh = Q + (size_t)bh * S_ * D_;
    const _Float16* Kbh = K + (size_t)bh * S_ * D_;
    const _Float16* Vbh = V + (size_t)bh * S_ * D_;

    const int arow = lane & 15;
    const int ncol = lane & 15;
    const int mh   = (lane >> 4) * 8;

    // Q fragments: 16x64 = two A frags (K-steps of 32), kept resident
    v16h qa[2];
    #pragma unroll
    for (int st = 0; st < 2; ++st)
        #pragma unroll
        for (int e = 0; e < 16; ++e)
            qa[st][e] = Qbh[(qbase + arow) * D_ + st * 32 + a_kmap(lane, e)];

    v8f zero = {};
    v8f O[4];
    #pragma unroll
    for (int nt = 0; nt < 4; ++nt) O[nt] = zero;
    float mrow[8], lrow[8];
    #pragma unroll
    for (int r = 0; r < 8; ++r) { mrow[r] = -__builtin_inff(); lrow[r] = 0.0f; }

    for (int kb = 0; kb < qbase + 16; kb += 32) {
        __syncthreads();
#if TDM_OK
        tdm_load_tile(lds_off_of(Klds), &Kbh[(size_t)kb * D_], 3u, 512u, 1u, 512u);
        tdm_load_tile(lds_off_of(Vlds), &Vbh[(size_t)kb * D_], 3u, 512u, 1u, 512u);
        __builtin_amdgcn_s_wait_tensorcnt(0);
#else
        for (int i = lane; i < 32 * D_; i += 32) {
            Klds[i] = Kbh[(size_t)kb * D_ + i];
            Vlds[i] = Vbh[(size_t)kb * D_ + i];
        }
#endif
        __syncthreads();
        // scores: two 16x16 tiles, each K-dim 64 = 2 chained WMMAs
        v8f s[2];
        #pragma unroll
        for (int t = 0; t < 2; ++t) {
            s[t] = zero;
            #pragma unroll
            for (int st = 0; st < 2; ++st) {
                v16h bf;   // B(k,col) = K[key=col][k]  (K^T), from LDS
                #pragma unroll
                for (int e = 0; e < 16; ++e)
                    bf[e] = Klds[(t * 16 + ncol) * D_ + st * 32 + b_kmap(lane, e)];
                s[t] = wmma_f16(qa[st], bf, s[t]);
            }
        }
        // scale + causal mask (element (m,n): m = mh + r, n = ncol)
        #pragma unroll
        for (int t = 0; t < 2; ++t)
            #pragma unroll
            for (int r = 0; r < 8; ++r) {
                const int key = kb + t * 16 + ncol;
                const int qi  = qbase + mh + r;
                const float v = s[t][r] * 0.125f;   // 1/sqrt(64)
                s[t][r] = (key <= qi) ? v : -__builtin_inff();
            }
        // row max over 32 keys: per-lane pair max, then 16-lane tree reduce
        float alpha[8];
        #pragma unroll
        for (int r = 0; r < 8; ++r) {
            float v = fmaxf(s[0][r], s[1][r]);
            v = fmaxf(v, __shfl_xor(v, 1, 32));
            v = fmaxf(v, __shfl_xor(v, 2, 32));
            v = fmaxf(v, __shfl_xor(v, 4, 32));
            v = fmaxf(v, __shfl_xor(v, 8, 32));
            const float mn = fmaxf(mrow[r], v);
            alpha[r] = expf(mrow[r] - mn);
            mrow[r]  = mn;
        }
        // exponentiate + row-sum reduce, update l, rescale O
        #pragma unroll
        for (int r = 0; r < 8; ++r) {
            s[0][r] = expf(s[0][r] - mrow[r]);
            s[1][r] = expf(s[1][r] - mrow[r]);
            float v = s[0][r] + s[1][r];
            v += __shfl_xor(v, 1, 32);
            v += __shfl_xor(v, 2, 32);
            v += __shfl_xor(v, 4, 32);
            v += __shfl_xor(v, 8, 32);
            lrow[r] = lrow[r] * alpha[r] + v;
        }
        #pragma unroll
        for (int nt = 0; nt < 4; ++nt)
            #pragma unroll
            for (int r = 0; r < 8; ++r) O[nt][r] *= alpha[r];
        // P (16x32) to LDS in row-major f16, then reload as A fragment
        #pragma unroll
        for (int t = 0; t < 2; ++t)
            #pragma unroll
            for (int r = 0; r < 8; ++r)
                Plds[(mh + r) * 32 + t * 16 + ncol] = (_Float16)s[t][r];
        __syncthreads();
        v16h pa;
        #pragma unroll
        for (int e = 0; e < 16; ++e)
            pa[e] = Plds[arow * 32 + a_kmap(lane, e)];
        // O (16x64) += P (16x32) @ V (32x64), V from LDS
        #pragma unroll
        for (int nt = 0; nt < 4; ++nt) {
            v16h bf;
            #pragma unroll
            for (int e = 0; e < 16; ++e)
                bf[e] = Vlds[b_kmap(lane, e) * D_ + nt * 16 + ncol];
            O[nt] = wmma_f16(pa, bf, O[nt]);
        }
    }
    // normalize rows and store, heads pre-concatenated: attn[b, s, h*64 + d]
    #pragma unroll
    for (int nt = 0; nt < 4; ++nt)
        #pragma unroll
        for (int r = 0; r < 8; ++r) {
            const float v = O[nt][r] / lrow[r];
            attn[((size_t)(b * S_ + qbase + mh + r)) * (H_ * D_) +
                 h * D_ + nt * 16 + ncol] = (_Float16)v;
        }
}

// ---------------------------------------------------------------------------
// Kernel 3: output projection  out = attn (4096x1024) @ Wo (1024x1024) + bo
// grid (B*S/64, E/64), block 128 (4 waves). A tile (64 rows x 64 B) and
// Wo tile (32 rows x 256 B) staged in LDS via TDM.
// ---------------------------------------------------------------------------
__global__ void mha_out_kernel(const _Float16* __restrict__ A,
                               const float* __restrict__ Wo,
                               const float* __restrict__ bo,
                               float* __restrict__ out) {
    __shared__ _Float16 Alds[64 * 32];
    __shared__ float    Wlds[32 * 64];
    const int tid   = threadIdx.x;
    const int lane  = tid & 31;
    const int wave  = tid >> 5;
    const int mrow0 = blockIdx.x * 64;
    const int mbase = mrow0 + wave * 16;
    const int nbase = blockIdx.y * 64;
    const int ncol  = lane & 15;
    const int arow  = lane & 15;
    const int KD    = H_ * D_;   // 1024

    v8f zero = {};
    v8f acc[4];
    #pragma unroll
    for (int nt = 0; nt < 4; ++nt) acc[nt] = zero;

    for (int k0 = 0; k0 < KD; k0 += 32) {
        __syncthreads();
#if TDM_OK
        if (wave == 0) {
            tdm_load_tile(lds_off_of(Alds), &A[(size_t)mrow0 * KD + k0],
                          3u, 8u, 64u, 256u);    // 64 rows x 64 B, stride 2 KB
            tdm_load_tile(lds_off_of(Wlds), &Wo[(size_t)k0 * E_ + nbase],
                          2u, 64u, 32u, (unsigned)E_);   // 32 rows x 64 f32
            __builtin_amdgcn_s_wait_tensorcnt(0);
        }
#else
        for (int i = tid; i < 64 * 32; i += 128)
            Alds[i] = A[(size_t)(mrow0 + (i >> 5)) * KD + k0 + (i & 31)];
        for (int i = tid; i < 32 * 64; i += 128)
            Wlds[i] = Wo[(size_t)(k0 + (i >> 6)) * E_ + nbase + (i & 63)];
#endif
        __syncthreads();
        v16h a;
        #pragma unroll
        for (int e = 0; e < 16; ++e)
            a[e] = Alds[(wave * 16 + arow) * 32 + a_kmap(lane, e)];
        #pragma unroll
        for (int nt = 0; nt < 4; ++nt) {
            v16h bf;
            #pragma unroll
            for (int e = 0; e < 16; ++e)
                bf[e] = (_Float16)Wlds[b_kmap(lane, e) * 64 + nt * 16 + ncol];
            acc[nt] = wmma_f16(a, bf, acc[nt]);
        }
    }
    const int mh = (lane >> 4) * 8;
    #pragma unroll
    for (int nt = 0; nt < 4; ++nt) {
        const int col = nbase + nt * 16 + ncol;
        const float bias = bo[col];
        #pragma unroll
        for (int r = 0; r < 8; ++r)
            out[(size_t)(mbase + mh + r) * E_ + col] = acc[nt][r] + bias;
    }
}

// ---------------------------------------------------------------------------
extern "C" void kernel_launch(void* const* d_in, const int* in_sizes, int n_in,
                              void* d_out, int out_size, void* d_ws, size_t ws_size,
                              hipStream_t stream) {
    (void)in_sizes; (void)n_in; (void)out_size; (void)ws_size;
    const float* x  = (const float*)d_in[0];
    const float* Wq = (const float*)d_in[1];
    const float* Wk = (const float*)d_in[2];
    const float* Wv = (const float*)d_in[3];
    const float* Wo = (const float*)d_in[4];
    const float* bo = (const float*)d_in[5];
    float* out = (float*)d_out;

    const size_t NQ = (size_t)B_ * H_ * S_ * D_;      // 4 Mi halves each
    _Float16* Qh = (_Float16*)d_ws;
    _Float16* Kh = Qh + NQ;
    _Float16* Vh = Kh + NQ;
    _Float16* At = Vh + NQ;                           // total 32 MB of d_ws

    dim3 gproj(S_ / 64, B_ * H_);
    mha_proj_kernel<<<gproj, 128, 0, stream>>>(x, Wq, Qh);
    mha_proj_kernel<<<gproj, 128, 0, stream>>>(x, Wk, Kh);
    mha_proj_kernel<<<gproj, 128, 0, stream>>>(x, Wv, Vh);
    mha_attn_kernel<<<dim3(S_ / 16, B_ * H_), 32, 0, stream>>>(Qh, Kh, Vh, At);
    mha_out_kernel<<<dim3(B_ * S_ / 64, E_ / 64), 128, 0, stream>>>(At, Wo, bo, out);
}
